// LVCBlock_83562883711360
// MI455X (gfx1250) — matrix operand order repaired and checked by reading, not compiled, not run on hardware
//
#include <hip/hip_runtime.h>

// CDNA5 (gfx1250) wave32 WMMA types
typedef __attribute__((ext_vector_type(16))) __bf16 v16bf;
typedef __attribute__((ext_vector_type(8)))  float  v8f;

static __device__ __forceinline__ float lrelu_f(float v) { return v >= 0.0f ? v : 0.2f * v; }

#define WMMA_BF16(A, B, C) \
  __builtin_amdgcn_wmma_f32_16x16x32_bf16(false, (A), false, (B), (short)0, (C), false, false)

// ---------------------------------------------------------------------------
// A-fragment packing (ISA 7.12.2, 16x32 bf16 tile):
//   lane<16 : elem 0-7 -> K=+0..7,  elem 8-15 -> K=+16..23, M=lane
//   lane>=16: elem 0-7 -> K=+8..15, elem 8-15 -> K=+24..31, M=lane-16
// Packed: A[(mt*KC+kc)*512 + lane*16 + e].  grid: x=KC*512/256 threads, y=mt.
// ---------------------------------------------------------------------------
template <int KK>
__global__ void __launch_bounds__(256)
pack_w_kernel(const float* __restrict__ W, __bf16* __restrict__ A, int Co, int Ci)
{
  const int Kdim = Ci * KK;
  const int KC = (Kdim + 31) >> 5;
  int tid  = blockIdx.x * 256 + threadIdx.x;   // exact: KC*512 threads
  int e    = tid & 15;
  int lane = (tid >> 4) & 31;
  int kc   = tid >> 9;
  int mt   = blockIdx.y;
  int row  = mt * 16 + (lane & 15);
  int kv   = kc * 32 + ((e < 8) ? 0 : 16) + ((lane < 16) ? 0 : 8) + (e & 7);
  int ci   = kv / KK;                           // constant divisor -> mulhi
  int j    = kv - ci * KK;
  int cic  = (ci < Ci) ? ci : Ci - 1;
  float v  = W[((size_t)row * Ci + cic) * KK + j];
  v = (row < Co && kv < Kdim) ? v : 0.0f;
  A[((size_t)mt * KC + kc) * 512 + (tid & 511)] = (__bf16)v;
}

// ---------------------------------------------------------------------------
// im2col -> B-fragment packing (branchless clamp+select, optional fused lrelu).
// B fragment: lane n<16 holds K=+0..7,+16..23 of col n; lane n+16 holds K=+8..15,+24..31.
// Packed: B[((b*NT+nt)*KC+kc)*512 + lane*16 + e].  grid: x=KC*512/256, y=nt, z=b.
// ---------------------------------------------------------------------------
template <int KK>
__global__ void __launch_bounds__(256)
pack_im2col_kernel(const float* __restrict__ X, __bf16* __restrict__ Bp,
                   int Ci, int L, int NT, int dil, int in_act)
{
  const int Kdim = Ci * KK;
  const int KC = (Kdim + 31) >> 5;
  int tid  = blockIdx.x * 256 + threadIdx.x;   // exact: KC*512 threads
  int e    = tid & 15;
  int lane = (tid >> 4) & 31;
  int kc   = tid >> 9;
  int nt   = blockIdx.y;
  int b    = blockIdx.z;
  int col  = nt * 16 + (lane & 15);
  int kv   = kc * 32 + ((e < 8) ? 0 : 16) + ((lane < 16) ? 0 : 8) + (e & 7);
  int ci   = kv / KK;
  int j    = kv - ci * KK;
  const int pad = ((KK - 1) / 2) * dil;
  int pos  = col + j * dil - pad;
  int posc = pos < 0 ? 0 : (pos >= L ? L - 1 : pos);
  int cic  = (ci < Ci) ? ci : Ci - 1;
  float v  = X[((size_t)b * Ci + cic) * L + posc];
  v = (kv < Kdim && pos >= 0 && pos < L) ? v : 0.0f;
  if (in_act) v = lrelu_f(v);
  Bp[(((size_t)b * NT + nt) * KC + kc) * 512 + (tid & 511)] = (__bf16)v;
}

// ---------------------------------------------------------------------------
// Pack predicted LVC kernels (layer slice of ko) into A-fragments per (b,s).
// A[co, ci*3+j], mrow = ((layer*64+ci)*128+co)*3+j.
// Packed: A[(((b*64+s)*8+mt)*6+kc)*512 + lane*16+e]. grid: x=12, y=s*8+mt, z=b.
// ---------------------------------------------------------------------------
__global__ void __launch_bounds__(256)
pack_lvc_kernel(const float* __restrict__ ko, __bf16* __restrict__ A, int layer)
{
  int tid  = blockIdx.x * 256 + threadIdx.x;   // exact: 6*512 = 3072
  int e    = tid & 15;
  int lane = (tid >> 4) & 31;
  int kc   = tid >> 9;
  int mt   = blockIdx.y & 7;
  int s    = blockIdx.y >> 3;
  int b    = blockIdx.z;
  int co   = mt * 16 + (lane & 15);                                         // 0..127
  int kv   = kc * 32 + ((e < 8) ? 0 : 16) + ((lane < 16) ? 0 : 8) + (e & 7); // 0..191
  int ci   = kv / 3;
  int j    = kv - ci * 3;
  int mrow = ((layer * 64 + ci) * 128 + co) * 3 + j;
  A[(((size_t)(b * 64 + s) * 8 + mt) * 6 + kc) * 512 + (tid & 511)] =
      (__bf16)ko[((size_t)b * 98304 + mrow) * 64 + s];
}

// ---------------------------------------------------------------------------
// GEMM on pre-packed fragments: one wave -> 32x32 output (2 M-tiles x 2 N-tiles,
// 4 accumulators).  Inner loop: 4 contiguous fragment loads + 4 WMMAs.
// grid: x = L/32, y = Co/32, z = batch;  blockDim = 32 (one wave).
// ---------------------------------------------------------------------------
__global__ void __launch_bounds__(32)
gemm_wmma_bf16(const __bf16* __restrict__ Apack, const __bf16* __restrict__ Bpack,
               const float* __restrict__ bias, const float* __restrict__ resid,
               float* __restrict__ Y, int Co, int L, int KC, int out_act)
{
  const int lane = threadIdx.x;
  const int nt0 = blockIdx.x * 2;
  const int mt0 = blockIdx.y * 2;
  const int b   = blockIdx.z;
  const int NT  = L >> 4;
  const size_t tstride = (size_t)KC * 32;       // v16bf units per 16-tile
  const v16bf* ap = (const v16bf*)Apack + (size_t)mt0 * tstride + lane;
  const v16bf* bp = (const v16bf*)Bpack + ((size_t)b * NT + nt0) * tstride + lane;

  v8f acc00 = {}, acc01 = {}, acc10 = {}, acc11 = {};
  for (int kc = 0; kc < KC; ++kc) {
    v16bf a0 = ap[(size_t)kc * 32];
    v16bf a1 = ap[(size_t)kc * 32 + tstride];
    v16bf b0 = bp[(size_t)kc * 32];
    v16bf b1 = bp[(size_t)kc * 32 + tstride];
    acc00 = WMMA_BF16(a0, b0, acc00);
    acc01 = WMMA_BF16(a0, b1, acc01);
    acc10 = WMMA_BF16(a1, b0, acc10);
    acc11 = WMMA_BF16(a1, b1, acc11);
  }
  // C/D layout: VGPR r, lane l -> M = r + 8*(l>=16), N = l&15
  const int rowadd = (lane >= 16) ? 8 : 0;
  const int c0 = nt0 * 16 + (lane & 15);
#pragma unroll
  for (int mi = 0; mi < 2; ++mi) {
    const int m0 = (mt0 + mi) * 16;
#pragma unroll
    for (int r = 0; r < 8; ++r) {
      int row = m0 + r + rowadd;
      float bb = bias[row];
      float v0 = (mi == 0 ? acc00[r] : acc10[r]) + bb;
      float v1 = (mi == 0 ? acc01[r] : acc11[r]) + bb;
      if (out_act) { v0 = lrelu_f(v0); v1 = lrelu_f(v1); }
      size_t o0 = ((size_t)b * Co + row) * L + c0;
      if (resid) { v0 += resid[o0]; v1 += resid[o0 + 16]; }
      Y[o0] = v0;
      Y[o0 + 16] = v1;
    }
  }
}

// ---------------------------------------------------------------------------
// ConvTranspose1d(C,C,16,stride 8,pad 4) fused with leaky-ReLU, branchless.
// ---------------------------------------------------------------------------
__global__ void __launch_bounds__(256)
convt_lrelu_kernel(const float* __restrict__ x, const float* __restrict__ w,
                   const float* __restrict__ bias, float* __restrict__ y)
{
  int idx = blockIdx.x * blockDim.x + threadIdx.x;
  int t  = idx & 16383;
  int co = (idx >> 14) & 63;
  int b  = idx >> 20;
  int tp = t + 4;
  int r  = tp & 7;
  int s1 = tp >> 3;
  float m1 = (s1 < 2048) ? 1.0f : 0.0f;
  int  s1c = (s1 < 2048) ? s1 : 2047;
  float m0 = (s1 >= 1) ? 1.0f : 0.0f;
  int  s0c = (s1 >= 1) ? s1 - 1 : 0;
  float acc = bias[co];
  const float* xb = x + (size_t)b * 64 * 2048;
#pragma unroll 4
  for (int ci = 0; ci < 64; ++ci) {
    const float* wr = w + ((size_t)ci * 64 + co) * 16;  // torch (Ci,Co,16) layout
    acc += lrelu_f(m1 * xb[ci * 2048 + s1c]) * wr[r];
    acc += lrelu_f(m0 * xb[ci * 2048 + s0c]) * wr[r + 8];
  }
  y[idx] = acc;
}

// ---------------------------------------------------------------------------
// Location-variable conv + gate, fused.  Per (b,s): o = K(b,s) @ im2col(act).
// Each wave: sigmoid tile (rows co) AND tanh tile (rows co+64) share one B
// fragment -> 3 fragment loads + 2 WMMAs per chunk; gate fused in epilogue.
// grid: x = s*16+nt (1024), y = co-pair tile (4), z = b (4); blockDim = 32.
// ---------------------------------------------------------------------------
__global__ void __launch_bounds__(32)
lvc_gate_wmma(const __bf16* __restrict__ Apack, const __bf16* __restrict__ Bpack,
              const float* __restrict__ bo, const float* __restrict__ xin,
              float* __restrict__ xout, int layer)
{
  const int lane = threadIdx.x;
  const int nt = blockIdx.x & 15;
  const int s  = blockIdx.x >> 4;
  const int cp = blockIdx.y;
  const int b  = blockIdx.z;
  const v16bf* ap = (const v16bf*)Apack + (size_t)(b * 64 + s) * 48 * 32 + lane;
  const v16bf* bp = (const v16bf*)Bpack + ((size_t)b * 1024 + s * 16 + nt) * 6 * 32 + lane;

  v8f accS = {}, accT = {};
  for (int kc = 0; kc < 6; ++kc) {
    v16bf aS = ap[((size_t)cp * 6 + kc) * 32];        // rows co      (sigmoid)
    v16bf aT = ap[((size_t)(cp + 4) * 6 + kc) * 32];  // rows co + 64 (tanh)
    v16bf bf = bp[(size_t)kc * 32];
    accS = WMMA_BF16(aS, bf, accS);
    accT = WMMA_BF16(aT, bf, accT);
  }
  const int rowadd = (lane >= 16) ? 8 : 0;
  const int t = s * 256 + nt * 16 + (lane & 15);
#pragma unroll
  for (int r = 0; r < 8; ++r) {
    int co = cp * 16 + r + rowadd;                    // 0..63
    float os = accS[r] + bo[((size_t)b * 512 + layer * 128 + co) * 64 + s];
    float ot = accT[r] + bo[((size_t)b * 512 + layer * 128 + 64 + co) * 64 + s];
    float g  = 1.0f / (1.0f + __expf(-os));
    size_t xi = ((size_t)b * 64 + co) * 16384 + t;
    xout[xi] = xin[xi] + g * tanhf(ot);
  }
}

// ---------------------------------------------------------------------------
extern "C" void kernel_launch(void* const* d_in, const int* in_sizes, int n_in,
                              void* d_out, int out_size, void* d_ws, size_t ws_size,
                              hipStream_t stream)
{
  (void)in_sizes; (void)n_in; (void)out_size; (void)ws_size;
  // setup_inputs() insertion order
  const float* X       = (const float*)d_in[0];
  const float* Cc      = (const float*)d_in[1];
  const float* convt_w = (const float*)d_in[2];
  const float* convt_b = (const float*)d_in[3];
  const float *amp_w1[3], *amp_b1[3], *amp_w2[3], *amp_b2[3];
  for (int a = 0; a < 3; ++a) {
    amp_w1[a] = (const float*)d_in[4 + 4 * a];
    amp_b1[a] = (const float*)d_in[5 + 4 * a];
    amp_w2[a] = (const float*)d_in[6 + 4 * a];
    amp_b2[a] = (const float*)d_in[7 + 4 * a];
  }
  const float* kp_in_w = (const float*)d_in[16];
  const float* kp_in_b = (const float*)d_in[17];
  const float *res_w1[3], *res_b1[3], *res_w2[3], *res_b2[3];
  for (int r = 0; r < 3; ++r) {
    res_w1[r] = (const float*)d_in[18 + 4 * r];
    res_b1[r] = (const float*)d_in[19 + 4 * r];
    res_w2[r] = (const float*)d_in[20 + 4 * r];
    res_b2[r] = (const float*)d_in[21 + 4 * r];
  }
  const float* kp_kw = (const float*)d_in[30];
  const float* kp_kb = (const float*)d_in[31];
  const float* kp_bw = (const float*)d_in[32];
  const float* kp_bb = (const float*)d_in[33];
  const float *blk_w[4], *blk_b[4];
  for (int i = 0; i < 4; ++i) {
    blk_w[i] = (const float*)d_in[34 + 2 * i];
    blk_b[i] = (const float*)d_in[35 + 2 * i];
  }

  // ---- workspace carve-out ----
  size_t off = 0;
  auto carve = [&](size_t bytes) -> void* {
    void* p = (char*)d_ws + off;
    off += (bytes + 255) & ~(size_t)255;
    return p;
  };
  float*  xbuf  = (float*)carve(sizeof(float) * 4ull * 64 * 16384);
  float*  tbuf  = (float*)carve(sizeof(float) * 4ull * 64 * 16384);
  float*  hbuf  = (float*)carve(sizeof(float) * 4ull * 64 * 64);
  float*  h2    = (float*)carve(sizeof(float) * 4ull * 64 * 64);
  float*  ko    = (float*)carve(sizeof(float) * 4ull * 98304 * 64);
  float*  bo    = (float*)carve(sizeof(float) * 4ull * 512 * 64);
  __bf16* lvcA  = (__bf16*)carve(sizeof(__bf16) * 4ull * 64 * 8 * 6 * 512);
  __bf16* bpBig = (__bf16*)carve(sizeof(__bf16) * 4ull * 1024 * 6 * 512);  // L=16384 im2col
  __bf16* bpSml = (__bf16*)carve(sizeof(__bf16) * 4ull * 4 * 40 * 512);    // L=64 im2col

  auto packw = [&](const float* W, int Co, int Ci, int kk) -> __bf16* {
    int KC = (Ci * kk + 31) / 32, MT = Co / 16;
    __bf16* A = (__bf16*)carve(sizeof(__bf16) * (size_t)MT * KC * 512);
    dim3 g((KC * 512) / 256, MT, 1);
    if (kk == 3) pack_w_kernel<3><<<g, dim3(256), 0, stream>>>(W, A, Co, Ci);
    else         pack_w_kernel<5><<<g, dim3(256), 0, stream>>>(W, A, Co, Ci);
    return A;
  };
  auto packB = [&](const float* Xp, __bf16* Bp, int Ci, int L, int kk, int dil, int ia) {
    int KC = (Ci * kk + 31) / 32, NT = L / 16;
    dim3 g((KC * 512) / 256, NT, 4);
    if (kk == 3) pack_im2col_kernel<3><<<g, dim3(256), 0, stream>>>(Xp, Bp, Ci, L, NT, dil, ia);
    else         pack_im2col_kernel<5><<<g, dim3(256), 0, stream>>>(Xp, Bp, Ci, L, NT, dil, ia);
  };
  auto gemm = [&](const __bf16* A, const __bf16* Bp, const float* bias, const float* resid,
                  float* Yp, int Co, int L, int KC, int oa) {
    dim3 g(L / 32, Co / 32, 4);
    gemm_wmma_bf16<<<g, dim3(32), 0, stream>>>(A, Bp, bias, resid, Yp, Co, L, KC, oa);
  };

  // pack static conv weights -> bf16 A-fragments
  __bf16 *ampA1[3], *ampA2[3], *resA1[3], *resA2[3], *blkA[4];
  for (int a = 0; a < 3; ++a) { ampA1[a] = packw(amp_w1[a], 64, 64, 3);
                                ampA2[a] = packw(amp_w2[a], 64, 64, 3); }
  __bf16* kpInA = packw(kp_in_w, 64, 256, 5);
  for (int r = 0; r < 3; ++r) { resA1[r] = packw(res_w1[r], 64, 64, 3);
                                resA2[r] = packw(res_w2[r], 64, 64, 3); }
  __bf16* kwA = packw(kp_kw, 98304, 64, 3);
  __bf16* bwA = packw(kp_bw, 512, 64, 3);
  for (int i = 0; i < 4; ++i) blkA[i] = packw(blk_w[i], 64, 64, 3);

  // 1) convt_pre: lrelu -> ConvTranspose1d (2048 -> 16384)
  convt_lrelu_kernel<<<dim3((4 * 64 * 16384) / 256), dim3(256), 0, stream>>>(
      X, convt_w, convt_b, xbuf);

  // 2) AMP residual stack: x += conv2(lrelu(conv1(lrelu(x), dil)))
  const int amp_dil[3] = {1, 3, 5};
  for (int a = 0; a < 3; ++a) {
    packB(xbuf, bpBig, 64, 16384, 3, amp_dil[a], 1);
    gemm(ampA1[a], bpBig, amp_b1[a], nullptr, tbuf, 64, 16384, 6, 0);
    packB(tbuf, bpBig, 64, 16384, 3, 1, 1);
    gemm(ampA2[a], bpBig, amp_b2[a], xbuf, xbuf, 64, 16384, 6, 0);
  }

  // 3) kernel predictor (on c, L=64)
  packB(Cc, bpSml, 256, 64, 5, 1, 0);
  gemm(kpInA, bpSml, kp_in_b, nullptr, hbuf, 64, 64, 40, 1);
  for (int r = 0; r < 3; ++r) {
    packB(hbuf, bpSml, 64, 64, 3, 1, 0);
    gemm(resA1[r], bpSml, res_b1[r], nullptr, h2, 64, 64, 6, 1);
    packB(h2, bpSml, 64, 64, 3, 1, 0);
    gemm(resA2[r], bpSml, res_b2[r], hbuf, hbuf, 64, 64, 6, 1);
  }
  packB(hbuf, bpSml, 64, 64, 3, 1, 0);              // shared by kp_kw and kp_bw
  gemm(kwA, bpSml, kp_kb, nullptr, ko, 98304, 64, 6, 0);  // dominant GEMM
  gemm(bwA, bpSml, kp_bb, nullptr, bo, 512, 64, 6, 0);

  // 4) LVC layers: dilated conv -> per-segment WMMA GEMM + fused gate
  const int dils[4] = {1, 3, 9, 27};
  for (int i = 0; i < 4; ++i) {
    packB(xbuf, bpBig, 64, 16384, 3, dils[i], 1);
    gemm(blkA[i], bpBig, blk_b[i], nullptr, tbuf, 64, 16384, 6, 1);
    packB(tbuf, bpBig, 64, 16384, 3, 1, 0);         // im2col of `out` (K=3, pad=1)
    pack_lvc_kernel<<<dim3(12, 512, 4), dim3(256), 0, stream>>>(ko, lvcA, i);
    float* xo = (i == 3) ? (float*)d_out : xbuf;    // last layer writes d_out
    lvc_gate_wmma<<<dim3(1024, 4, 4), dim3(32), 0, stream>>>(lvcA, bpBig, bo, xbuf, xo, i);
  }
}
